// Multi_CrossAttention_89455578841879
// MI455X (gfx1250) — matrix operand order repaired and checked by reading, not compiled, not run on hardware
//
#include <hip/hip_runtime.h>
#include <hip/hip_bf16.h>
#include <cstdint>

// Problem constants (match reference)
#define BATCH   4
#define SQ      2048
#define SK      2048
#define HID     512
#define HEADS   8
#define HDIM    64
#define MTOT    (BATCH * SQ)        // 8192 rows for projections

typedef _Float16 v16h __attribute__((ext_vector_type(16)));
typedef _Float16 v8h  __attribute__((ext_vector_type(8)));
typedef _Float16 v4h  __attribute__((ext_vector_type(4)));
typedef float    v8f  __attribute__((ext_vector_type(8)));

union V16U { v16h v; v8h h[2]; };

static __device__ __forceinline__ v8f wmma_f16(v16h a, v16h b, v8f c) {
    // D = A(16x32 f16) * B(32x16 f16) + C(16x16 f32)
    return __builtin_amdgcn_wmma_f32_16x16x32_f16(
        /*neg_a=*/false, a, /*neg_b=*/false, b,
        /*c_mod=*/(short)0, c, /*reuse_a=*/false, /*reuse_b=*/false);
}

// A fragment (16x32, M x K): lane L -> m = L%16, hg = L/16
//   a[i]   = A[m][hg*16 + i]       i=0..7
//   a[8+i] = A[m][hg*16 + 8 + i]
static __device__ __forceinline__ v16h load_a_frag(const _Float16* row, int hg) {
    V16U u;
    u.h[0] = *(const v8h*)(row + hg * 16);
    u.h[1] = *(const v8h*)(row + hg * 16 + 8);
    return u.v;
}

// B fragment (32x16, K x N) from a transposed tile Bt[n][k] (contiguous in k):
//   lane L -> n = L%16, b[i] = B[hg*16 + i][n] = Bt[n][hg*16 + i]
static __device__ __forceinline__ v16h load_b_frag(const _Float16* p) {
    V16U u;
    u.h[0] = *(const v8h*)(p);
    u.h[1] = *(const v8h*)(p + 8);
    return u.v;
}

static __device__ __forceinline__ float rmax16(float v) {
    #pragma unroll
    for (int m = 1; m < 16; m <<= 1) v = fmaxf(v, __shfl_xor(v, m, 32));
    return v;
}
static __device__ __forceinline__ float rsum16(float v) {
    #pragma unroll
    for (int m = 1; m < 16; m <<= 1) v += __shfl_xor(v, m, 32);
    return v;
}

// CDNA5 async global->LDS copy (ASYNCcnt-tracked). LDS aperture keeps the
// byte offset in addr[31:0], so truncating the generic pointer is the LDS addr.
static __device__ __forceinline__ void async_b128(const void* gptr, const void* lptr) {
    unsigned loff = (unsigned)(uintptr_t)lptr;
    asm volatile("global_load_async_to_lds_b128 %0, %1, off"
                 :: "v"(loff), "v"(gptr) : "memory");
}
static __device__ __forceinline__ void wait_async0() {
    asm volatile("s_wait_asynccnt 0x0" ::: "memory");
}
static __device__ __forceinline__ void wait_ds0() {
    asm volatile("s_wait_dscnt 0x0" ::: "memory");
}

// ---------------------------------------------------------------------------
// Projection GEMM: Out(f16)[M,512] = X(f32)[M,512] @ W(f32)[512,512]
// Block tile 128(M) x 64(N), 8 waves each do 16x64. K staged 32/iter in LDS.
// ---------------------------------------------------------------------------
__global__ __launch_bounds__(256) void proj_kernel(const float* __restrict__ X,
                                                   const float* __restrict__ W,
                                                   _Float16* __restrict__ Out) {
    __shared__ _Float16 Xl[128][40];  // 128 rows x 32 k (+8 pad)
    __shared__ _Float16 Wt[64][40];   // 64 n x 32 k (transposed, +8 pad)

    const int tid  = threadIdx.x;
    const int wave = tid >> 5;
    const int lane = tid & 31;
    const int lm   = lane & 15;
    const int hg   = lane >> 4;
    const int m0   = blockIdx.x * 128;
    const int n0   = blockIdx.y * 64;

    v8f acc[4];
    #pragma unroll
    for (int t = 0; t < 4; ++t)
        #pragma unroll
        for (int r = 0; r < 8; ++r) acc[t][r] = 0.0f;

    for (int kc = 0; kc < HID; kc += 32) {
        __syncthreads();
        #pragma unroll
        for (int rr = 0; rr < 4; ++rr) {
            const int row = rr * 32 + (tid >> 3);
            const int c4  = (tid & 7) * 4;
            const float4 xv = *(const float4*)(X + (size_t)(m0 + row) * HID + kc + c4);
            v4h hv; hv[0] = (_Float16)xv.x; hv[1] = (_Float16)xv.y;
            hv[2] = (_Float16)xv.z; hv[3] = (_Float16)xv.w;
            *(v4h*)(&Xl[row][c4]) = hv;
        }
        {
            const int kk = tid >> 3;
            const int nn = (tid & 7) * 8;
            const float4 w0 = *(const float4*)(W + (size_t)(kc + kk) * HID + n0 + nn);
            const float4 w1 = *(const float4*)(W + (size_t)(kc + kk) * HID + n0 + nn + 4);
            Wt[nn + 0][kk] = (_Float16)w0.x; Wt[nn + 1][kk] = (_Float16)w0.y;
            Wt[nn + 2][kk] = (_Float16)w0.z; Wt[nn + 3][kk] = (_Float16)w0.w;
            Wt[nn + 4][kk] = (_Float16)w1.x; Wt[nn + 5][kk] = (_Float16)w1.y;
            Wt[nn + 6][kk] = (_Float16)w1.z; Wt[nn + 7][kk] = (_Float16)w1.w;
        }
        __syncthreads();

        const v16h a = load_a_frag(&Xl[wave * 16 + lm][0], hg);
        #pragma unroll
        for (int t = 0; t < 4; ++t) {
            const v16h b = load_b_frag(&Wt[t * 16 + lm][hg * 16]);
            acc[t] = wmma_f16(a, b, acc[t]);
        }
    }

    #pragma unroll
    for (int t = 0; t < 4; ++t)
        #pragma unroll
        for (int r = 0; r < 8; ++r) {
            const int row = m0 + wave * 16 + r + 8 * hg;
            const int col = n0 + t * 16 + lm;
            Out[(size_t)row * HID + col] = (_Float16)acc[t][r];
        }
}

// ---------------------------------------------------------------------------
// Same GEMM but writes the result TRANSPOSED per head: OutT[B][H][D][SK].
// Used for V so the flash kernel can async-copy row-major (d-major) V tiles.
// ---------------------------------------------------------------------------
__global__ __launch_bounds__(256) void projT_kernel(const float* __restrict__ X,
                                                    const float* __restrict__ W,
                                                    _Float16* __restrict__ OutT) {
    __shared__ _Float16 Xl[128][40];
    __shared__ _Float16 Wt[64][40];

    const int tid  = threadIdx.x;
    const int wave = tid >> 5;
    const int lane = tid & 31;
    const int lm   = lane & 15;
    const int hg   = lane >> 4;
    const int m0   = blockIdx.x * 128;
    const int n0   = blockIdx.y * 64;   // one head per y-block (HDIM==64)

    v8f acc[4];
    #pragma unroll
    for (int t = 0; t < 4; ++t)
        #pragma unroll
        for (int r = 0; r < 8; ++r) acc[t][r] = 0.0f;

    for (int kc = 0; kc < HID; kc += 32) {
        __syncthreads();
        #pragma unroll
        for (int rr = 0; rr < 4; ++rr) {
            const int row = rr * 32 + (tid >> 3);
            const int c4  = (tid & 7) * 4;
            const float4 xv = *(const float4*)(X + (size_t)(m0 + row) * HID + kc + c4);
            v4h hv; hv[0] = (_Float16)xv.x; hv[1] = (_Float16)xv.y;
            hv[2] = (_Float16)xv.z; hv[3] = (_Float16)xv.w;
            *(v4h*)(&Xl[row][c4]) = hv;
        }
        {
            const int kk = tid >> 3;
            const int nn = (tid & 7) * 8;
            const float4 w0 = *(const float4*)(W + (size_t)(kc + kk) * HID + n0 + nn);
            const float4 w1 = *(const float4*)(W + (size_t)(kc + kk) * HID + n0 + nn + 4);
            Wt[nn + 0][kk] = (_Float16)w0.x; Wt[nn + 1][kk] = (_Float16)w0.y;
            Wt[nn + 2][kk] = (_Float16)w0.z; Wt[nn + 3][kk] = (_Float16)w0.w;
            Wt[nn + 4][kk] = (_Float16)w1.x; Wt[nn + 5][kk] = (_Float16)w1.y;
            Wt[nn + 6][kk] = (_Float16)w1.z; Wt[nn + 7][kk] = (_Float16)w1.w;
        }
        __syncthreads();

        const v16h a = load_a_frag(&Xl[wave * 16 + lm][0], hg);
        #pragma unroll
        for (int t = 0; t < 4; ++t) {
            const v16h b = load_b_frag(&Wt[t * 16 + lm][hg * 16]);
            acc[t] = wmma_f16(a, b, acc[t]);
        }
    }

    const int h = n0 >> 6;
    #pragma unroll
    for (int t = 0; t < 4; ++t)
        #pragma unroll
        for (int r = 0; r < 8; ++r) {
            const int row = m0 + wave * 16 + r + 8 * hg;   // b*SQ + s
            const int bb  = row >> 11;                     // row / SQ
            const int s   = row & (SK - 1);
            const int d   = t * 16 + lm;
            OutT[(((size_t)bb * HEADS + h) * HDIM + d) * SK + s] = (_Float16)acc[t][r];
        }
}

// ---------------------------------------------------------------------------
// Flash attention, 64-key blocks, double-buffered async global->LDS staging.
// Block = (b, h, 128 q rows); 8 waves x 16 q rows each.
// ---------------------------------------------------------------------------
__global__ __launch_bounds__(256) void flash_kernel(const _Float16* __restrict__ Q,
                                                    const _Float16* __restrict__ K,
                                                    const _Float16* __restrict__ VT,
                                                    const int* __restrict__ mask,
                                                    _Float16* __restrict__ Ctx) {
    __shared__ _Float16 Kl[2][64][72];   // keys x d, row major   (18 KB)
    __shared__ _Float16 Vl[2][64][72];   // d x keys (VT layout)  (18 KB)
    __shared__ _Float16 Pl[8][16][72];   // per-wave P: 16 q x 64 keys (18 KB)

    const int tid  = threadIdx.x;
    const int wave = tid >> 5;
    const int lane = tid & 31;
    const int lm   = lane & 15;
    const int hg   = lane >> 4;
    const int b    = blockIdx.z;
    const int h    = blockIdx.y;
    const int qw   = blockIdx.x * 128 + wave * 16;

    // Persistent Q A-fragments (d = 0..31, 32..63)
    const _Float16* qrow = Q + (size_t)(b * SQ + qw + lm) * HID + h * HDIM;
    const v16h aq0 = load_a_frag(qrow, hg);
    const v16h aq1 = load_a_frag(qrow + 32, hg);

    v8f acc[4];
    float lsum[8], mrow[8];
    #pragma unroll
    for (int t = 0; t < 4; ++t)
        #pragma unroll
        for (int r = 0; r < 8; ++r) acc[t][r] = 0.0f;
    #pragma unroll
    for (int r = 0; r < 8; ++r) { lsum[r] = 0.0f; mrow[r] = -3.0e38f; }

    const float scale = 0.125f;          // 1/sqrt(HDIM)
    const int srow = tid >> 2;           // 0..63 (staging row)
    const int sq16 = (tid & 3) * 16;     // 0..48 (staging col, halfs)

    auto stage = [&](int buf, int k0) {
        const _Float16* kg = K + (size_t)(b * SK + k0 + srow) * HID + h * HDIM + sq16;
        async_b128(kg,     &Kl[buf][srow][sq16]);
        async_b128(kg + 8, &Kl[buf][srow][sq16 + 8]);
        const _Float16* vg = VT + (((size_t)b * HEADS + h) * HDIM + srow) * SK + k0 + sq16;
        async_b128(vg,     &Vl[buf][srow][sq16]);
        async_b128(vg + 8, &Vl[buf][srow][sq16 + 8]);
    };

    stage(0, 0);

    for (int kb = 0; kb < SK / 64; ++kb) {
        const int k0  = kb * 64;
        const int buf = kb & 1;

        wait_async0();        // own async loads for `buf` complete
        __syncthreads();      // all waves' loads visible; prior reads of buf^1 done
        if (kb + 1 < SK / 64) stage(buf ^ 1, k0 + 64);

        // Scores: 4 n-tiles of 16 keys; d=64 -> 2 WMMAs each
        v8f s[4];
        #pragma unroll
        for (int t = 0; t < 4; ++t) {
            v8f c;
            #pragma unroll
            for (int r = 0; r < 8; ++r) c[r] = 0.0f;
            c = wmma_f16(aq0, load_b_frag(&Kl[buf][t * 16 + lm][hg * 16]), c);
            c = wmma_f16(aq1, load_b_frag(&Kl[buf][t * 16 + lm][32 + hg * 16]), c);
            s[t] = c;
        }

        // Mask (mask != 0 -> -1e9) then scale
        #pragma unroll
        for (int t = 0; t < 4; ++t)
            #pragma unroll
            for (int r = 0; r < 8; ++r) {
                const size_t mi = ((size_t)b * SQ + (qw + r + 8 * hg)) * SK + (k0 + t * 16 + lm);
                const int mv = mask[mi];
                float sv = (mv != 0) ? -1.0e9f : s[t][r];
                s[t][r] = sv * scale;
            }

        // Online softmax per q row
        #pragma unroll
        for (int r = 0; r < 8; ++r) {
            const float rm = rmax16(fmaxf(fmaxf(s[0][r], s[1][r]), fmaxf(s[2][r], s[3][r])));
            const float mnew = fmaxf(mrow[r], rm);
            const float corr = __expf(mrow[r] - mnew);
            mrow[r] = mnew;
            float p[4], ps = 0.0f;
            #pragma unroll
            for (int t = 0; t < 4; ++t) { p[t] = __expf(s[t][r] - mnew); ps += p[t]; }
            lsum[r] = lsum[r] * corr + rsum16(ps);
            #pragma unroll
            for (int t = 0; t < 4; ++t) acc[t][r] *= corr;
            #pragma unroll
            for (int t = 0; t < 4; ++t)
                Pl[wave][r + 8 * hg][t * 16 + lm] = (_Float16)p[t];
        }
        wait_ds0();   // within-wave P C-layout -> A-layout round trip

        // P(16x64) @ V(64x64): 2 k-chunks x 4 d-tiles
        #pragma unroll
        for (int c = 0; c < 2; ++c) {
            const v16h pa = load_a_frag(&Pl[wave][lm][c * 32], hg);
            #pragma unroll
            for (int t = 0; t < 4; ++t) {
                const v16h bv = load_b_frag(&Vl[buf][t * 16 + lm][c * 32 + hg * 16]);
                acc[t] = wmma_f16(pa, bv, acc[t]);
            }
        }
    }

    // Normalize and write ctx (f16) at [b, q, h*64 + d]
    #pragma unroll
    for (int t = 0; t < 4; ++t)
        #pragma unroll
        for (int r = 0; r < 8; ++r) {
            const int row = qw + r + 8 * hg;
            const int d   = t * 16 + lm;
            const float v = acc[t][r] / lsum[r];
            Ctx[(size_t)(b * SQ + row) * HID + h * HDIM + d] = (_Float16)v;
        }
}

// ---------------------------------------------------------------------------
// Output projection: Out(f32)[M,512] = Ctx(f16)[M,512] @ Wo(f32)[512,512] + bo
// Ctx tile staged with async global->LDS (already f16).
// ---------------------------------------------------------------------------
__global__ __launch_bounds__(256) void oproj_kernel(const _Float16* __restrict__ Ctx,
                                                    const float* __restrict__ W,
                                                    const float* __restrict__ bias,
                                                    float* __restrict__ Out) {
    __shared__ _Float16 Xl[128][40];
    __shared__ _Float16 Wt[64][40];

    const int tid  = threadIdx.x;
    const int wave = tid >> 5;
    const int lane = tid & 31;
    const int lm   = lane & 15;
    const int hg   = lane >> 4;
    const int m0   = blockIdx.x * 128;
    const int n0   = blockIdx.y * 64;

    v8f acc[4];
    #pragma unroll
    for (int t = 0; t < 4; ++t)
        #pragma unroll
        for (int r = 0; r < 8; ++r) acc[t][r] = 0.0f;

    for (int kc = 0; kc < HID; kc += 32) {
        __syncthreads();
        // Async-stage the f16 ctx tile: thread -> (row = tid/2, 16-half chunk)
        {
            const int row = tid >> 1;
            const int c16 = (tid & 1) * 16;
            const _Float16* cg = Ctx + (size_t)(m0 + row) * HID + kc + c16;
            async_b128(cg,     &Xl[row][c16]);
            async_b128(cg + 8, &Xl[row][c16 + 8]);
        }
        // Manual (converting) stage of the weight tile, transposed
        {
            const int kk = tid >> 3;
            const int nn = (tid & 7) * 8;
            const float4 w0 = *(const float4*)(W + (size_t)(kc + kk) * HID + n0 + nn);
            const float4 w1 = *(const float4*)(W + (size_t)(kc + kk) * HID + n0 + nn + 4);
            Wt[nn + 0][kk] = (_Float16)w0.x; Wt[nn + 1][kk] = (_Float16)w0.y;
            Wt[nn + 2][kk] = (_Float16)w0.z; Wt[nn + 3][kk] = (_Float16)w0.w;
            Wt[nn + 4][kk] = (_Float16)w1.x; Wt[nn + 5][kk] = (_Float16)w1.y;
            Wt[nn + 6][kk] = (_Float16)w1.z; Wt[nn + 7][kk] = (_Float16)w1.w;
        }
        wait_async0();
        __syncthreads();

        const v16h a = load_a_frag(&Xl[wave * 16 + lm][0], hg);
        #pragma unroll
        for (int t = 0; t < 4; ++t) {
            const v16h b = load_b_frag(&Wt[t * 16 + lm][hg * 16]);
            acc[t] = wmma_f16(a, b, acc[t]);
        }
    }

    #pragma unroll
    for (int t = 0; t < 4; ++t) {
        const float bv = bias[n0 + t * 16 + lm];
        #pragma unroll
        for (int r = 0; r < 8; ++r) {
            const int row = m0 + wave * 16 + r + 8 * hg;
            const int col = n0 + t * 16 + lm;
            Out[(size_t)row * HID + col] = acc[t][r] + bv;
        }
    }
}

// ---------------------------------------------------------------------------
extern "C" void kernel_launch(void* const* d_in, const int* in_sizes, int n_in,
                              void* d_out, int out_size, void* d_ws, size_t ws_size,
                              hipStream_t stream) {
    (void)in_sizes; (void)n_in; (void)out_size; (void)ws_size;

    const float* x    = (const float*)d_in[0];
    const float* y    = (const float*)d_in[1];
    const int*   msk  = (const int*)d_in[2];
    const float* Wq   = (const float*)d_in[3];
    const float* Wk   = (const float*)d_in[4];
    const float* Wv   = (const float*)d_in[5];
    const float* Wo   = (const float*)d_in[6];
    const float* bo   = (const float*)d_in[7];
    float*       out  = (float*)d_out;

    const size_t QSZ = (size_t)MTOT * HID;  // elements per f16 tensor
    _Float16* Qw  = (_Float16*)d_ws;
    _Float16* Kw  = Qw + QSZ;
    _Float16* VwT = Kw + QSZ;   // [B][H][D][SK]
    _Float16* Cw  = VwT + QSZ;

    dim3 gproj(MTOT / 128, HID / 64);       // 64 x 8
    proj_kernel <<<gproj, 256, 0, stream>>>(x, Wq, Qw);
    proj_kernel <<<gproj, 256, 0, stream>>>(y, Wk, Kw);
    projT_kernel<<<gproj, 256, 0, stream>>>(y, Wv, VwT);

    dim3 gfa(SQ / 128, HEADS, BATCH);       // 16 x 8 x 4
    flash_kernel<<<gfa, 256, 0, stream>>>(Qw, Kw, VwT, msk, Cw);

    oproj_kernel<<<gproj, 256, 0, stream>>>(Cw, Wo, bo, out);
}
